// VQVAE_17540646437249
// MI455X (gfx1250) — matrix operand order repaired and checked by reading, not compiled, not run on hardware
//
#include <hip/hip_runtime.h>
#include <hip/hip_bf16.h>

#define B_DIM 4096
#define T_DIM 64
#define K_DIM 512
#define D_DIM 64
#define KC    256          // codes staged per LDS chunk (32 KB bf16)
#define BETA  0.25f

typedef __attribute__((ext_vector_type(16))) __bf16 v16bf;
typedef __attribute__((ext_vector_type(8)))  float  v8f;

union FragA {
    v16bf v;
    uint4 q[2];
    unsigned short h[16];
};

static __device__ inline unsigned short f32_to_bf16(float f) {
    union { float f; unsigned int u; } c; c.f = f;
    unsigned int r = 0x7FFFu + ((c.u >> 16) & 1u);   // round-to-nearest-even
    return (unsigned short)((c.u + r) >> 16);
}

// ---------------------------------------------------------------------------
// Kernel 1: store bf16(-2*c) into ws, compute ||c||^2 (fp32), zero flags.
// grid: T*K/8 blocks of 256 (8 waves, one code-row each; lane handles 2 elems)
// ---------------------------------------------------------------------------
__global__ void vq_prep(const float* __restrict__ cb,
                        unsigned short* __restrict__ cb16,
                        float* __restrict__ cnorm,
                        float* __restrict__ used) {
    const int tid  = threadIdx.x;
    const int lane = tid & 31;
    const int wv   = tid >> 5;
    const int row  = blockIdx.x * 8 + wv;            // 0 .. T*K-1
    const float2 v = ((const float2*)(cb + (size_t)row * D_DIM))[lane];
    // pre-scale by -2 so WMMA produces ||c||^2 - 2*c.z directly (C = cnorm)
    unsigned int packed = (unsigned int)f32_to_bf16(-2.0f * v.x) |
                          ((unsigned int)f32_to_bf16(-2.0f * v.y) << 16);
    ((unsigned int*)cb16)[(size_t)row * (D_DIM / 2) + lane] = packed;
    float p = fmaf(v.x, v.x, v.y * v.y);
#pragma unroll
    for (int off = 16; off > 0; off >>= 1) p += __shfl_xor(p, off, 32);
    if (lane == 0) cnorm[row] = p;
    if (tid < 8) used[blockIdx.x * 8 + tid] = 0.0f;  // zero utilization flags
}

// ---------------------------------------------------------------------------
// Kernel 2: distances via bf16 WMMA (C seeded with ||c||^2), argmin, gather.
// grid: (B/64, T), 128 threads (4 waves x 16 z-rows each)
// ---------------------------------------------------------------------------
__global__ void __launch_bounds__(128)
vq_main(const float* __restrict__ z_e,
        const float* __restrict__ cb,                 // fp32 codebooks (gather)
        const unsigned short* __restrict__ cb16,      // bf16(-2*c)
        const float* __restrict__ cnorm,
        float* __restrict__ used,
        float* __restrict__ partial,                  // [4096 * 4]
        float* __restrict__ zq,
        float* __restrict__ tokens) {
    __shared__ __align__(16) unsigned short lds_cb[KC * D_DIM];  // 32 KB

    const int t    = blockIdx.y;
    const int lane = threadIdx.x & 31;
    const int wave = threadIdx.x >> 5;
    const int nrow = lane & 15;   // z-row within wave tile / A code-row in tile
    const int half = lane >> 4;   // K-half (A/B striping) and code-half (C)
    const int rowb = blockIdx.x * 64 + wave * 16 + nrow;

    // ---- B fragments (z rows as columns) + partial ||z||^2 (fp32) ----
    FragA bfr[2];
    float znp = 0.f;
    const float* zbase = z_e + ((size_t)rowb * T_DIM + t) * D_DIM;
#pragma unroll
    for (int s = 0; s < 2; ++s) {
        const float* zp = zbase + s * 32 + half * 8;
#pragma unroll
        for (int q = 0; q < 4; ++q) {
            float4 f = ((const float4*)zp)[q + (q & 2)];
            znp = fmaf(f.x, f.x, znp); znp = fmaf(f.y, f.y, znp);
            znp = fmaf(f.z, f.z, znp); znp = fmaf(f.w, f.w, znp);
            bfr[s].h[q * 4 + 0] = f32_to_bf16(f.x);
            bfr[s].h[q * 4 + 1] = f32_to_bf16(f.y);
            bfr[s].h[q * 4 + 2] = f32_to_bf16(f.z);
            bfr[s].h[q * 4 + 3] = f32_to_bf16(f.w);
        }
    }

    // two independent running-min chains to shorten the dependency chain
    float mv0 = 3.402823466e+38f, mv1 = 3.402823466e+38f;
    int   mi0 = 0,                mi1 = 0;

    for (int chunk = 0; chunk < K_DIM / KC; ++chunk) {
        __syncthreads();
        {   // stage 256 codes x 64 d of bf16(-2c) into LDS
            const uint4* src = (const uint4*)(cb16 + ((size_t)t * K_DIM + chunk * KC) * D_DIM);
            uint4* dst = (uint4*)lds_cb;
            for (int i = threadIdx.x; i < KC * D_DIM / 8; i += 128) dst[i] = src[i];
        }
        __syncthreads();

        const float4* cn4   = (const float4*)(cnorm + (size_t)t * K_DIM + chunk * KC);
        const uint4*  ldsu4 = (const uint4*)lds_cb;

        for (int tile = 0; tile < KC / 16; ++tile) {
            const int rowu4 = (tile * 16 + nrow) * (D_DIM / 8);
            FragA a0, a1;                      // 16-bit A 16x32 layout (ISA 7.12.2)
            a0.q[0] = ldsu4[rowu4 + half];         // K =  k½*8 .. +7
            a0.q[1] = ldsu4[rowu4 + 2 + half];     // K = 16+k½*8 .. +7
            a1.q[0] = ldsu4[rowu4 + 4 + half];     // second k-step (d 32..63)
            a1.q[1] = ldsu4[rowu4 + 6 + half];

            // seed accumulator with ||c||^2 for this lane's 8 codes
            float4 cna = cn4[tile * 4 + half * 2];
            float4 cnb = cn4[tile * 4 + half * 2 + 1];
            v8f acc = {cna.x, cna.y, cna.z, cna.w, cnb.x, cnb.y, cnb.z, cnb.w};

            acc = __builtin_amdgcn_wmma_f32_16x16x32_bf16(
                false, a0.v, false, bfr[0].v, (short)0, acc, false, false);
            acc = __builtin_amdgcn_wmma_f32_16x16x32_bf16(
                false, a1.v, false, bfr[1].v, (short)0, acc, false, false);
            // acc[j] == ||c_m||^2 - 2 c_m . z_n   (m = mbase+j, n = nrow)

            const int mbase = chunk * KC + tile * 16 + half * 8;
#pragma unroll
            for (int j = 0; j < 4; ++j) {
                float va = acc[j];
                float vb = acc[4 + j];
                if (va < mv0) { mv0 = va; mi0 = mbase + j; }
                if (vb < mv1) { mv1 = vb; mi1 = mbase + 4 + j; }
            }
        }
    }

    // merge the two chains (lower index wins ties)
    float minv = mv0;
    int   mini = mi0;
    if (mv1 < minv || (mv1 == minv && mi1 < mini)) { minv = mv1; mini = mi1; }

    // merge the two code-halves (lane L <-> L^16); lower index wins ties
    {
        float v2 = __shfl_xor(minv, 16, 32);
        int   i2 = __shfl_xor(mini, 16, 32);
        if (v2 < minv || (v2 == minv && i2 < mini)) { minv = v2; mini = i2; }
        znp += __shfl_xor(znp, 16, 32);          // full ||z||^2 for this row
    }

    if (half == 0) {
        tokens[(size_t)rowb * T_DIM + t] = (float)mini;
        used[(size_t)t * K_DIM + mini]   = 1.0f;        // idempotent store
    }

    // loss partial: sum over this wave's 16 rows of ||z - c*||^2
    float contrib = (half == 0) ? (znp + minv) : 0.0f;
#pragma unroll
    for (int off = 16; off > 0; off >>= 1) contrib += __shfl_xor(contrib, off, 32);
    if (lane == 0) {
        const int bid = blockIdx.y * gridDim.x + blockIdx.x;
        partial[bid * 4 + wave] = contrib;
    }

    // gather selected codebook rows (exact fp32) into z_q_st
#pragma unroll 1
    for (int r = 0; r < 16; ++r) {
        const int code = __shfl(mini, r, 32);
        const float2* srcr = (const float2*)(cb + ((size_t)t * K_DIM + code) * D_DIM);
        float2* dstr = (float2*)(zq + ((size_t)(blockIdx.x * 64 + wave * 16 + r) * T_DIM + t) * D_DIM);
        dstr[lane] = srcr[lane];
    }
}

// ---------------------------------------------------------------------------
// Kernel 3: deterministic final reductions -> vq_loss, utilization
// ---------------------------------------------------------------------------
__global__ void vq_finalize(const float* __restrict__ used,
                            const float* __restrict__ partial,
                            float* __restrict__ out_loss,
                            float* __restrict__ out_util) {
    __shared__ float su[256];
    __shared__ float sl[256];
    const int tid = threadIdx.x;
    float u = 0.f, l = 0.f;
    for (int i = tid; i < T_DIM * K_DIM; i += 256) u += used[i];
    for (int i = tid; i < 4096 * 4;     i += 256) l += partial[i];
    su[tid] = u; sl[tid] = l;
    __syncthreads();
    for (int s = 128; s > 0; s >>= 1) {
        if (tid < s) { su[tid] += su[tid + s]; sl[tid] += sl[tid + s]; }
        __syncthreads();
    }
    if (tid == 0) {
        *out_util = su[0] / (float)(T_DIM * K_DIM);
        *out_loss = BETA * sl[0] / (float)((size_t)B_DIM * T_DIM * D_DIM);
    }
}

// ---------------------------------------------------------------------------
extern "C" void kernel_launch(void* const* d_in, const int* in_sizes, int n_in,
                              void* d_out, int out_size, void* d_ws, size_t ws_size,
                              hipStream_t stream) {
    const float* z_e = (const float*)d_in[0];      // (B, T, D) fp32
    const float* cb  = (const float*)d_in[1];      // (T, K, D) fp32

    char* ws = (char*)d_ws;
    unsigned short* cb16  = (unsigned short*)ws;                        // 4 MiB
    float* cnorm   = (float*)(ws + 4194304);                            // 128 KB
    float* used    = (float*)(ws + 4194304 + 131072);                   // 128 KB
    float* partial = (float*)(ws + 4194304 + 131072 + 131072);          // 64 KB

    float* out    = (float*)d_out;
    float* zq     = out;                                                // B*T*D
    float* tokens = out + (size_t)B_DIM * T_DIM * D_DIM;                // B*T
    float* loss   = tokens + (size_t)B_DIM * T_DIM;
    float* util   = loss + 1;

    vq_prep<<<dim3(T_DIM * K_DIM / 8), 256, 0, stream>>>(cb, cb16, cnorm, used);
    vq_main<<<dim3(B_DIM / 64, T_DIM), 128, 0, stream>>>(z_e, cb, cb16, cnorm,
                                                         used, partial, zq, tokens);
    vq_finalize<<<1, 256, 0, stream>>>(used, partial, loss, util);
}
